// MultiHeadAttention_71150428225958
// MI455X (gfx1250) — compile-verified, matrix operand order checked by reading, not run on hardware
//
#include <hip/hip_runtime.h>

typedef unsigned short u16;
typedef __attribute__((ext_vector_type(16))) __bf16 v16bf;
typedef __attribute__((ext_vector_type(8)))  float  v8f;

#define S_LEN 2048
#define D_DIM 1024
#define H_NUM 16
#define DH_NUM 64

__device__ __forceinline__ u16 f2bf(float f) {
    unsigned u = __float_as_uint(f);
    u += 0x7FFFu + ((u >> 16) & 1u);
    return (u16)(u >> 16);
}

struct Pack32 { uint4 a, b; };
struct P16 { u16 h[16]; };

// Load one bf16 WMMA operand fragment (A-pattern; B uses the same pattern with
// lane -> N).  Lane-half 0: K in [0,8)u[16,24); lane-half 1: K in [8,16)u[24,32).
// Caller passes p already offset by (row, basek); we read p[0..7] and p[16..23].
__device__ __forceinline__ v16bf ldfrag(const u16* p) {
    Pack32 s;
    s.a = *(const uint4*)(p);
    s.b = *(const uint4*)(p + 16);
    return __builtin_bit_cast(v16bf, s);
}

__device__ __forceinline__ v8f wmma_bf16(v16bf a, v16bf b, v8f c) {
    return __builtin_amdgcn_wmma_f32_16x16x32_bf16(false, a, false, b, (short)0, c,
                                                   false, false);
}

// ---------------------------------------------------------------- fp32 -> bf16
__global__ __launch_bounds__(256) void cvt_bf16_kernel(const float* __restrict__ in,
                                                       u16* __restrict__ out, int n) {
    int i = blockIdx.x * 256 + threadIdx.x;
    if (i < n) out[i] = f2bf(in[i]);
}

// ---------------------------------------------------------------- LayerNorm
__global__ __launch_bounds__(256) void ln_kernel(const float* __restrict__ x,
                                                 const float* __restrict__ g,
                                                 const float* __restrict__ bb,
                                                 u16* __restrict__ xn) {
    __shared__ float red[8];
    int tid = threadIdx.x, lane = tid & 31, wave = tid >> 5;
    size_t row = blockIdx.x;
    const float* xr = x + row * D_DIM;
    float v[4];
    float s = 0.f;
#pragma unroll
    for (int i = 0; i < 4; i++) { v[i] = xr[tid + i * 256]; s += v[i]; }
#pragma unroll
    for (int m = 16; m >= 1; m >>= 1) s += __shfl_xor(s, m);
    if (lane == 0) red[wave] = s;
    __syncthreads();
    float tot = 0.f;
#pragma unroll
    for (int i = 0; i < 8; i++) tot += red[i];
    float mean = tot * (1.f / (float)D_DIM);
    __syncthreads();
    float s2 = 0.f;
#pragma unroll
    for (int i = 0; i < 4; i++) { float d = v[i] - mean; s2 += d * d; }
#pragma unroll
    for (int m = 16; m >= 1; m >>= 1) s2 += __shfl_xor(s2, m);
    if (lane == 0) red[wave] = s2;
    __syncthreads();
    float tot2 = 0.f;
#pragma unroll
    for (int i = 0; i < 8; i++) tot2 += red[i];
    float rstd = rsqrtf(tot2 * (1.f / (float)D_DIM) + 1e-5f);
#pragma unroll
    for (int i = 0; i < 4; i++) {
        int c = tid + i * 256;
        xn[row * D_DIM + c] = f2bf((v[i] - mean) * rstd * g[c] + bb[c]);
    }
}

// ---------------------------------------------------------------- Tiled WMMA GEMM
// C[M,N] = A[M,K] * W[N,K]^T (+ epilogue).  BM=128 BN=128 BK=32, 8 waves,
// wave tile 64x32 (4x2 WMMA accumulators).
// MODE 0: QKV — bias from (b0|b1), split-store bf16 q(*0.125)/k/v.
// MODE 1: out-proj — fp32 out = acc + b0[col] + resid.
template <int MODE>
__global__ __launch_bounds__(256) void gemm_kernel(
    const u16* __restrict__ A, const u16* __restrict__ W, int K_dim,
    const float* __restrict__ b0, const float* __restrict__ b1,
    u16* __restrict__ qb, u16* __restrict__ kb, u16* __restrict__ vb,
    const float* __restrict__ resid, float* __restrict__ outp) {
    __shared__ u16 As[128][48];
    __shared__ u16 Bs[128][48];
    int tid = threadIdx.x, lane = tid & 31, wave = tid >> 5;
    int wm = wave & 1, wn = wave >> 1;
    int half = lane >> 4, r = lane & 15, bk = half * 8;
    const size_t M0 = (size_t)blockIdx.y * 128;
    const size_t N0 = (size_t)blockIdx.x * 128;
    const u16* Aptr = A + M0 * K_dim;
    const u16* Wptr = W + N0 * K_dim;

    v8f acc[4][2] = {};
    for (int k0 = 0; k0 < K_dim; k0 += 32) {
        __syncthreads();
#pragma unroll
        for (int i = 0; i < 2; i++) {
            int c = tid + i * 256;
            int row = c >> 2, col = (c & 3) * 8;
            *(uint4*)&As[row][col] = *(const uint4*)(Aptr + (size_t)row * K_dim + k0 + col);
            *(uint4*)&Bs[row][col] = *(const uint4*)(Wptr + (size_t)row * K_dim + k0 + col);
        }
        __syncthreads();
        v16bf af[4], bfr[2];
#pragma unroll
        for (int i = 0; i < 4; i++) af[i] = ldfrag(&As[wm * 64 + i * 16 + r][bk]);
#pragma unroll
        for (int j = 0; j < 2; j++) bfr[j] = ldfrag(&Bs[wn * 32 + j * 16 + r][bk]);
#pragma unroll
        for (int i = 0; i < 4; i++)
#pragma unroll
            for (int j = 0; j < 2; j++) acc[i][j] = wmma_bf16(af[i], bfr[j], acc[i][j]);
    }

#pragma unroll
    for (int i = 0; i < 4; i++)
#pragma unroll
        for (int j = 0; j < 2; j++) {
            size_t rowb = M0 + wm * 64 + i * 16 + 8 * half;
            int col = (int)N0 + wn * 32 + j * 16 + r;
#pragma unroll
            for (int e = 0; e < 8; e++) {
                size_t rr = rowb + e;
                float vv = acc[i][j][e];
                if (MODE == 0) {
                    float bias = (col < D_DIM) ? b0[col] : b1[col - D_DIM];
                    float t = vv + bias;
                    if (col < D_DIM)
                        qb[rr * D_DIM + col] = f2bf(t * 0.125f);  // fold 1/sqrt(DH)
                    else if (col < 2 * D_DIM)
                        kb[rr * D_DIM + (col - D_DIM)] = f2bf(t);
                    else
                        vb[rr * D_DIM + (col - 2 * D_DIM)] = f2bf(t);
                } else {
                    outp[rr * D_DIM + col] = vv + b0[col] + resid[rr * D_DIM + col];
                }
            }
        }
}

// ---------------------------------------------------------------- V transpose: vt[bh][dh][s]
__global__ __launch_bounds__(256) void vtrans_kernel(const u16* __restrict__ v,
                                                     u16* __restrict__ vt) {
    __shared__ u16 t[64][65];
    int tid = threadIdx.x;
    int bh = blockIdx.y, b = bh >> 4, h = bh & 15;
    int s0 = blockIdx.x * 64;
    int dh = tid & 63, sr = tid >> 6;
#pragma unroll
    for (int i = 0; i < 64; i += 4)
        t[sr + i][dh] = v[(size_t)(b * S_LEN + s0 + sr + i) * D_DIM + h * DH_NUM + dh];
    __syncthreads();
    int sc = tid & 63, dr = tid >> 6;
#pragma unroll
    for (int i = 0; i < 64; i += 4)
        vt[((size_t)bh * DH_NUM + dr + i) * S_LEN + s0 + sc] = t[sc][dr + i];
}

// ---------------------------------------------------------------- Flash attention
// grid (B*H, S/128); wave w owns queries [qw, qw+16).  All products are kept
// TRANSPOSED:  S^T = K * Q^T   and   O^T = V^T * P.  In the C-layout each lane
// then owns exactly ONE query (col = lane&15), so the online-softmax stats are
// lane scalars (one shfl_xor(16) merges the two key half-groups), and the S^T
// C-layout is bit-identical to the B-operand layout needed for V^T * P — P is
// packed in registers, no LDS staging at all.  All 16 fragment loads of an
// iteration are hoisted ahead of the WMMAs so they issue as one clause, and
// masking runs only on the (rare) diagonal / length-boundary blocks.
__global__ __launch_bounds__(256) void attn_kernel(const u16* __restrict__ q,
                                                   const u16* __restrict__ k,
                                                   const u16* __restrict__ vt,
                                                   const int* __restrict__ lens,
                                                   u16* __restrict__ ctx) {
    int tid = threadIdx.x, lane = tid & 31, wave = tid >> 5;
    int half = lane >> 4, r = lane & 15, bk = half * 8;
    int bh = blockIdx.x, b = bh >> 4, h = bh & 15;
    int qw = blockIdx.y * 128 + wave * 16;
    int len = lens[b];
    int myq = qw + r;                        // the single query this lane owns
    int lim = (myq + 1 < len) ? myq + 1 : len;  // key valid iff key < lim

    // Q as B-operand (lane r = query), both 32-wide dh k-steps.
    const u16* qbase = q + ((size_t)(b * S_LEN + qw + r) * D_DIM + h * DH_NUM) + bk;
    v16bf qf[2];
    qf[0] = ldfrag(qbase);
    qf[1] = ldfrag(qbase + 32);

    // Induction bases: K row r (advance by keys), V^T row r of dh-tile 0.
    const u16* kp0 = k + ((size_t)(b * S_LEN + r) * D_DIM + h * DH_NUM) + bk;
    const u16* vp0 = vt + ((size_t)(bh * DH_NUM + r) * S_LEN) + bk;

    float mrun = -1e30f, lrun = 0.f;
    v8f Ov[4] = {};

    int kmax = qw + 15;
    if (len - 1 < kmax) kmax = len - 1;

    for (int kbp = 0; kbp <= kmax; kbp += 32) {
        // ---- all 16 operand loads up front (one big clause, overlapped waits)
        const u16* ka = kp0 + (size_t)kbp * D_DIM;
        const u16* kc = ka + (size_t)16 * D_DIM;
        v16bf kf0 = ldfrag(ka);
        v16bf kf1 = ldfrag(ka + 32);
        v16bf kf2 = ldfrag(kc);
        v16bf kf3 = ldfrag(kc + 32);
        v16bf vf[4];
#pragma unroll
        for (int j = 0; j < 4; j++)
            vf[j] = ldfrag(vp0 + (size_t)j * 16 * S_LEN + kbp);

        // ---- S^T tiles: rows = keys (t*16 + e + 8*half), col = query
        v8f Sacc[2] = {};
        Sacc[0] = wmma_bf16(kf0, qf[0], Sacc[0]);
        Sacc[0] = wmma_bf16(kf1, qf[1], Sacc[0]);
        Sacc[1] = wmma_bf16(kf2, qf[0], Sacc[1]);
        Sacc[1] = wmma_bf16(kf3, qf[1], Sacc[1]);

        // ---- mask (only on boundary blocks) + per-lane online softmax
        float sv[2][8];
        float rowmax = -3.0e38f;
        bool full = (kbp + 32 <= qw + 1) && (kbp + 32 <= len);  // wave-uniform
        if (full) {
#pragma unroll
            for (int t = 0; t < 2; t++)
#pragma unroll
                for (int e = 0; e < 8; e++) {
                    sv[t][e] = Sacc[t][e];
                    rowmax = fmaxf(rowmax, sv[t][e]);
                }
        } else {
            int kbase0 = kbp + 8 * half;
#pragma unroll
            for (int t = 0; t < 2; t++)
#pragma unroll
                for (int e = 0; e < 8; e++) {
                    int key = kbase0 + t * 16 + e;
                    sv[t][e] = (key < lim) ? Sacc[t][e] : -3.0e38f;
                    rowmax = fmaxf(rowmax, sv[t][e]);
                }
        }
        rowmax = fmaxf(rowmax, __shfl_xor(rowmax, 16));
        float mnew = fmaxf(mrun, rowmax);
        float scale = __expf(mrun - mnew);
        float rsum = 0.f;
        P16 ph;
#pragma unroll
        for (int t = 0; t < 2; t++)
#pragma unroll
            for (int e = 0; e < 8; e++) {
                float p = __expf(sv[t][e] - mnew);
                rsum += p;
                ph.h[t * 8 + e] = f2bf(p);  // B-operand half m = t*8+e -> key t*16+e+8*half
            }
        rsum += __shfl_xor(rsum, 16);
        lrun = lrun * scale + rsum;
        mrun = mnew;
        v16bf pf = __builtin_bit_cast(v16bf, ph);

        // ---- O^T += V^T * P  (rows = dh, col = query)
#pragma unroll
        for (int j = 0; j < 4; j++) {
#pragma unroll
            for (int e = 0; e < 8; e++) Ov[j][e] *= scale;
            Ov[j] = wmma_bf16(vf[j], pf, Ov[j]);
        }
    }

    float inv = 1.f / lrun;
#pragma unroll
    for (int j = 0; j < 4; j++)
#pragma unroll
        for (int e = 0; e < 8; e++) {
            int dh = j * 16 + e + 8 * half;
            ctx[(size_t)(b * S_LEN + myq) * D_DIM + h * DH_NUM + dh] = f2bf(Ov[j][e] * inv);
        }
}

// ---------------------------------------------------------------- launch
extern "C" void kernel_launch(void* const* d_in, const int* in_sizes, int n_in,
                              void* d_out, int out_size, void* d_ws, size_t ws_size,
                              hipStream_t stream) {
    (void)in_sizes; (void)n_in; (void)out_size; (void)ws_size;
    const float* x    = (const float*)d_in[0];
    const int*   lens = (const int*)d_in[1];
    const float* Wq   = (const float*)d_in[2];
    const float* bq   = (const float*)d_in[3];
    const float* Wkv  = (const float*)d_in[4];
    const float* bkv  = (const float*)d_in[5];
    const float* Wo   = (const float*)d_in[6];
    const float* bo   = (const float*)d_in[7];
    const float* ln_g = (const float*)d_in[8];
    const float* ln_b = (const float*)d_in[9];
    float* out = (float*)d_out;

    const size_t BS = 4 * (size_t)S_LEN;            // 8192 rows
    const size_t DD = (size_t)D_DIM * D_DIM;        // 1M
    char* ws = (char*)d_ws;
    size_t off = 0;
    auto alloc = [&](size_t elems) { u16* p = (u16*)(ws + off); off += elems * sizeof(u16); return p; };
    u16* wqkv = alloc(3 * DD);        // [Wq ; Wkv] rows, bf16
    u16* wo   = alloc(DD);
    u16* xn   = alloc(BS * D_DIM);
    u16* qb   = alloc(BS * D_DIM);
    u16* kbuf = alloc(BS * D_DIM);
    u16* vbuf = alloc(BS * D_DIM);
    u16* vtb  = alloc(BS * D_DIM);
    u16* ctx  = alloc(BS * D_DIM);

    cvt_bf16_kernel<<<(int)(DD / 256), 256, 0, stream>>>(Wq, wqkv, (int)DD);
    cvt_bf16_kernel<<<(int)(2 * DD / 256), 256, 0, stream>>>(Wkv, wqkv + DD, (int)(2 * DD));
    cvt_bf16_kernel<<<(int)(DD / 256), 256, 0, stream>>>(Wo, wo, (int)DD);

    ln_kernel<<<(int)BS, 256, 0, stream>>>(x, ln_g, ln_b, xn);

    dim3 gqkv(3 * D_DIM / 128, (int)(BS / 128));   // (24, 64)
    gemm_kernel<0><<<gqkv, 256, 0, stream>>>(xn, wqkv, D_DIM, bq, bkv,
                                             qb, kbuf, vbuf, nullptr, nullptr);

    vtrans_kernel<<<dim3(S_LEN / 64, 4 * H_NUM), 256, 0, stream>>>(vbuf, vtb);

    attn_kernel<<<dim3(4 * H_NUM, S_LEN / 128), 256, 0, stream>>>(qb, kbuf, vtb, lens, ctx);

    dim3 gout(D_DIM / 128, (int)(BS / 128));       // (8, 64)
    gemm_kernel<1><<<gout, 256, 0, stream>>>(ctx, wo, D_DIM, bo, nullptr,
                                             nullptr, nullptr, nullptr, x, out);
}
